// SpatialSelfAttention_34368328303165
// MI455X (gfx1250) — matrix operand order repaired and checked by reading, not compiled, not run on hardware
//
#include <hip/hip_runtime.h>
#include <hip/hip_bf16.h>

// ---------------------------------------------------------------------------
// SpatialSelfAttention for MI455X (gfx1250, wave32, WMMA bf16 -> f32 accum)
// N=8, T=4096, D=H=128.  Compute-bound (~71 GFLOP vs ~50MB traffic).
// ---------------------------------------------------------------------------

typedef __attribute__((ext_vector_type(16))) __bf16 v16bf;
typedef __attribute__((ext_vector_type(8)))  __bf16 v8bf;
typedef __attribute__((ext_vector_type(8)))  float  v8f;

#define NB   8
#define TT   4096
#define DD   128

static __device__ __forceinline__ v16bf load_frag(const __bf16* lo, const __bf16* hi) {
  v8bf a = *(const v8bf*)lo;
  v8bf b = *(const v8bf*)hi;
  v16bf r;
#pragma unroll
  for (int i = 0; i < 8; ++i) { r[i] = a[i]; r[i + 8] = b[i]; }
  return r;
}

static __device__ __forceinline__ v8f wmma_bf16(v16bf a, v16bf b, v8f c) {
  // D = A(16x32 bf16) * B(32x16 bf16) + C(16x16 f32)
  return __builtin_amdgcn_wmma_f32_16x16x32_bf16(false, a, false, b, (short)0, c,
                                                 false, false);
}

static __device__ __forceinline__ float rmax16(float v) {
#pragma unroll
  for (int m = 1; m <= 8; m <<= 1) v = fmaxf(v, __shfl_xor(v, m, 32));
  return v;
}
static __device__ __forceinline__ float rsum16(float v) {
#pragma unroll
  for (int m = 1; m <= 8; m <<= 1) v += __shfl_xor(v, m, 32);
  return v;
}

// ---------------------------------------------------------------------------
// Kernel 1: Q = (H*Wq + bq)/64 ; K = H*Wk + bk  (bf16 out, row-major)
//           also HbT[n][d][t] = bf16(H[n][t][d])  for the P*H matmul.
// grid 512 x 128 threads (4 waves, 16 rows each -> 64 rows/block)
// ---------------------------------------------------------------------------
__global__ __launch_bounds__(128) void
proj_qk_kernel(const float* __restrict__ H, const float* __restrict__ Wq,
               const float* __restrict__ bq, const float* __restrict__ Wk,
               const float* __restrict__ bk, __bf16* __restrict__ Qb,
               __bf16* __restrict__ Kb, __bf16* __restrict__ HbT) {
  __shared__ __attribute__((aligned(16))) __bf16 WqT[DD * DD];
  __shared__ __attribute__((aligned(16))) __bf16 WkT[DD * DD];
  __shared__ float bql[DD];
  __shared__ float bkl[DD];

  const int tid  = threadIdx.x;
  const int lane = tid & 31;
  const int w    = tid >> 5;
  const int R0   = blockIdx.x * 64;  // global row (n*T + t) base; 4096 % 64 == 0

  // Stage W^T (bf16) + biases into LDS.
  for (int i = tid; i < DD * DD; i += 128) {
    int d = i >> 7, h = i & 127;
    WqT[h * DD + d] = (__bf16)Wq[i];
    WkT[h * DD + d] = (__bf16)Wk[i];
  }
  if (tid < DD) { bql[tid] = bq[tid]; bkl[tid] = bk[tid]; }

  // Emit transposed bf16 H while W stages.
  const int nIdx  = R0 >> 12;
  const int tBase = R0 & (TT - 1);
  for (int i = tid; i < 64 * DD; i += 128) {
    int r = i >> 7, d = i & 127;
    HbT[((size_t)(nIdx * DD + d)) * TT + tBase + r] =
        (__bf16)H[(size_t)(R0 + r) * DD + d];
  }
  __syncthreads();

  // A-fragments: 16 rows x 128 (k) per wave, ISA 16-bit A layout.
  const int   row  = R0 + 16 * w + (lane & 15);
  const float* hrp = H + (size_t)row * DD;
  v16bf Af[4];
#pragma unroll
  for (int c = 0; c < 4; ++c) {
    int kb = 32 * c + 8 * (lane >> 4);
    float4 f0 = *(const float4*)(hrp + kb);
    float4 f1 = *(const float4*)(hrp + kb + 4);
    float4 f2 = *(const float4*)(hrp + kb + 16);
    float4 f3 = *(const float4*)(hrp + kb + 20);
    Af[c][0]  = (__bf16)f0.x; Af[c][1]  = (__bf16)f0.y;
    Af[c][2]  = (__bf16)f0.z; Af[c][3]  = (__bf16)f0.w;
    Af[c][4]  = (__bf16)f1.x; Af[c][5]  = (__bf16)f1.y;
    Af[c][6]  = (__bf16)f1.z; Af[c][7]  = (__bf16)f1.w;
    Af[c][8]  = (__bf16)f2.x; Af[c][9]  = (__bf16)f2.y;
    Af[c][10] = (__bf16)f2.z; Af[c][11] = (__bf16)f2.w;
    Af[c][12] = (__bf16)f3.x; Af[c][13] = (__bf16)f3.y;
    Af[c][14] = (__bf16)f3.z; Af[c][15] = (__bf16)f3.w;
  }

  const float qscale = 0.015625f;  // 1/sqrt(T) baked into Q
#pragma unroll
  for (int n0 = 0; n0 < 8; ++n0) {
    v8f aq = {};
    v8f ak = {};
#pragma unroll
    for (int c = 0; c < 4; ++c) {
      const __bf16* pq = &WqT[(n0 * 16 + (lane & 15)) * DD + 32 * c + 8 * (lane >> 4)];
      const __bf16* pk = &WkT[(n0 * 16 + (lane & 15)) * DD + 32 * c + 8 * (lane >> 4)];
      aq = wmma_bf16(Af[c], load_frag(pq, pq + 16), aq);
      ak = wmma_bf16(Af[c], load_frag(pk, pk + 16), ak);
    }
    const int   col = n0 * 16 + (lane & 15);
    const float vbq = bql[col];
    const float vbk = bkl[col];
#pragma unroll
    for (int r = 0; r < 8; ++r) {
      int orow = R0 + 16 * w + r + 8 * (lane >> 4);
      Qb[(size_t)orow * DD + col] = (__bf16)((aq[r] + vbq) * qscale);
      Kb[(size_t)orow * DD + col] = (__bf16)(ak[r] + vbk);
    }
  }
}

// ---------------------------------------------------------------------------
// Kernel 2: flash attention.  out = softmax(Q K^T) * H + H
// grid (T/64, N) x 128 threads; wave handles 16 query rows; keys in 32-chunks.
// ---------------------------------------------------------------------------
__global__ __launch_bounds__(128) void
flash_attn_kernel(const float* __restrict__ Hf, const __bf16* __restrict__ Qb,
                  const __bf16* __restrict__ Kb, const __bf16* __restrict__ HbT,
                  float* __restrict__ out) {
  __shared__ __attribute__((aligned(16))) __bf16 Plds[4][16][32];

  const int    tid  = threadIdx.x;
  const int    lane = tid & 31;
  const int    w    = tid >> 5;
  const int    n    = blockIdx.y;
  const int    qrow = blockIdx.x * 64 + 16 * w;  // row within batch
  const size_t base = (size_t)n * TT * DD;

  // Q A-fragments for this wave's 16 rows.
  v16bf Qf[4];
  {
    const __bf16* qrp = Qb + base + (size_t)(qrow + (lane & 15)) * DD;
#pragma unroll
    for (int c = 0; c < 4; ++c) {
      int kb = 32 * c + 8 * (lane >> 4);
      Qf[c]  = load_frag(qrp + kb, qrp + kb + 16);
    }
  }

  v8f O[8];
#pragma unroll
  for (int j = 0; j < 8; ++j) { v8f z = {}; O[j] = z; }
  float mrow[8], lrow[8];
#pragma unroll
  for (int r = 0; r < 8; ++r) { mrow[r] = -1e30f; lrow[r] = 0.0f; }

  const __bf16* Kbp  = Kb + base;
  const __bf16* HTbp = HbT + (size_t)n * DD * TT;

  for (int s0 = 0; s0 < TT; s0 += 32) {
    // S = Q * K^T  (two 16x16 tiles covering keys s0..s0+31)
    v8f S0 = {};
    v8f S1 = {};
#pragma unroll
    for (int c = 0; c < 4; ++c) {
      int h0 = 32 * c + 8 * (lane >> 4);
      const __bf16* k0 = Kbp + (size_t)(s0 + (lane & 15)) * DD + h0;
      const __bf16* k1 = Kbp + (size_t)(s0 + 16 + (lane & 15)) * DD + h0;
      S0 = wmma_bf16(Qf[c], load_frag(k0, k0 + 16), S0);
      S1 = wmma_bf16(Qf[c], load_frag(k1, k1 + 16), S1);
    }

    // Online softmax update (row stats per C-fragment component).
#pragma unroll
    for (int r = 0; r < 8; ++r) {
      float cm = rmax16(fmaxf(S0[r], S1[r]));
      float mn = fmaxf(mrow[r], cm);
      float al = __expf(mrow[r] - mn);
      float p0 = __expf(S0[r] - mn);
      float p1 = __expf(S1[r] - mn);
      float rs = rsum16(p0 + p1);
      lrow[r]  = lrow[r] * al + rs;
      mrow[r]  = mn;
      int pr = r + 8 * (lane >> 4);
      Plds[w][pr][lane & 15]      = (__bf16)p0;
      Plds[w][pr][16 + (lane & 15)] = (__bf16)p1;
#pragma unroll
      for (int j = 0; j < 8; ++j) O[j][r] *= al;
    }

    // C-layout -> A-layout via per-wave LDS tile.
    __asm__ volatile("s_wait_dscnt 0" ::: "memory");
    const __bf16* pp = &Plds[w][lane & 15][8 * (lane >> 4)];
    v16bf Pf = load_frag(pp, pp + 16);

    // O += P * H   (B fragments contiguous thanks to HbT)
#pragma unroll
    for (int j = 0; j < 8; ++j) {
      const __bf16* hb =
          HTbp + (size_t)(16 * j + (lane & 15)) * TT + s0 + 8 * (lane >> 4);
      O[j] = wmma_bf16(Pf, load_frag(hb, hb + 16), O[j]);
    }
  }

  // Normalize + residual.
#pragma unroll
  for (int r = 0; r < 8; ++r) {
    float inv = 1.0f / lrow[r];
    int   t   = qrow + r + 8 * (lane >> 4);
#pragma unroll
    for (int j = 0; j < 8; ++j) {
      int    d   = 16 * j + (lane & 15);
      size_t idx = base + (size_t)t * DD + d;
      out[idx]   = O[j][r] * inv + Hf[idx];
    }
  }
}

// ---------------------------------------------------------------------------
extern "C" void kernel_launch(void* const* d_in, const int* in_sizes, int n_in,
                              void* d_out, int out_size, void* d_ws, size_t ws_size,
                              hipStream_t stream) {
  const float* H  = (const float*)d_in[0];
  const float* Wq = (const float*)d_in[1];
  const float* bq = (const float*)d_in[2];
  const float* Wk = (const float*)d_in[3];
  const float* bk = (const float*)d_in[4];
  float* out = (float*)d_out;

  const size_t elems = (size_t)NB * TT * DD;  // 4,194,304
  __bf16* Qb  = (__bf16*)d_ws;
  __bf16* Kb  = Qb + elems;
  __bf16* HbT = Kb + elems;  // total 24 MB of scratch

  proj_qk_kernel<<<dim3((NB * TT) / 64), dim3(128), 0, stream>>>(
      H, Wq, bq, Wk, bk, Qb, Kb, HbT);
  flash_attn_kernel<<<dim3(TT / 64, NB), dim3(128), 0, stream>>>(
      H, Qb, Kb, HbT, out);
}
